// MultiHeaded_4080218931880
// MI455X (gfx1250) — compile-verified, matrix operand order checked by reading, not run on hardware
//
#include <hip/hip_runtime.h>

// ---------------------------------------------------------------------------
// MultiHeaded attention for MI455X (gfx1250, wave32, WMMA bf16 + TDM)
// B=2, F=T=2048, C=1024, H=16, D=64.  All matrix math via
// v_wmma_f32_16x16x32_bf16 (f32 accumulate).  Flash-attention style fused
// softmax so the 2048x2048 score matrix never touches HBM.  Double-buffered
// LDS staging: V via the Tensor Data Mover (async, TENSORcnt), K via
// register-transposed global loads, both overlapped with the WMMA stream.
// ---------------------------------------------------------------------------

typedef __attribute__((ext_vector_type(16))) __bf16 v16bf;
typedef __attribute__((ext_vector_type(8)))  float  v8f;

#define B_SZ   2
#define F_LEN  2048
#define T_LEN  2048
#define C_DIM  1024
#define NH     16
#define D_HEAD 64
#define HD     (NH * D_HEAD)                 // 1024
#define BATCH_STRIDE (HD * F_LEN)            // 2097152
#define MASK_BSTRIDE (F_LEN * T_LEN)

__device__ __forceinline__ unsigned short f2bf(float f) {
  unsigned int u = __float_as_uint(f);
  u += 0x7FFFu + ((u >> 16) & 1u);           // round-to-nearest-even
  return (unsigned short)(u >> 16);
}

__device__ __forceinline__ __bf16 bits2bf(unsigned short u) {
  return __builtin_bit_cast(__bf16, u);
}

// CDNA5 16-bit A/B fragment K mapping (CDNA5 ISA):
// lanes 0-15 hold K = {0..7, 16..23}; lanes 16-31 hold K = {8..15, 24..31}
__device__ __forceinline__ int klocal(int j, int hi) {
  return j + ((j >= 8) ? 8 : 0) + hi * 8;
}

__device__ __forceinline__ v8f wmma_bf16(v16bf a, v16bf b, v8f c) {
  return __builtin_amdgcn_wmma_f32_16x16x32_bf16(
      false, a, false, b, (short)0, c, false, false);
}

// ---------------------------------------------------------------------------
// Tensor Data Mover (CDNA5): 2D tile Global -> LDS with LDS row padding.
// Descriptor layout per CDNA5 ISA.
// ---------------------------------------------------------------------------
#if defined(__gfx1250__) && __has_builtin(__builtin_amdgcn_tensor_load_to_lds)
#define HAVE_TDM 1
typedef __attribute__((ext_vector_type(4))) unsigned int u32x4;
typedef __attribute__((ext_vector_type(8))) int i32x8;
typedef __attribute__((ext_vector_type(4))) int i32x4;

__device__ __forceinline__ unsigned lds_addr_of(const void* p) {
  return (unsigned)(unsigned long)(__attribute__((address_space(3))) const void*)p;
}

// tile_w in 4-byte units, tile_h rows, row stride in 4-byte units.
// pad_int_enc: pad after 2^(enc+1) dwords; pad_amt_enc: (enc+1) dwords pad.
__device__ __forceinline__ void tdm_load_2d(
    const void* gptr, unsigned lds_byte, unsigned tile_w_u32, unsigned tile_h,
    unsigned stride_u32, unsigned pad_int_enc, unsigned pad_amt_enc) {
  unsigned long ga = (unsigned long)gptr;
  u32x4 g0;
  g0[0] = 1u;                                             // count=1 (valid, user)
  g0[1] = lds_byte;                                       // lds_addr
  g0[2] = (unsigned)(ga & 0xFFFFFFFFu);                   // global_addr[31:0]
  g0[3] = (unsigned)((ga >> 32) & 0x1FFFFFFu) | (2u << 30); // addr[56:32]|type=2
  const unsigned td0 = 0x40000000u, td1 = 0x40000000u;    // huge dims: no OOB clip
  i32x8 g1;
  g1[0] = (int)((2u << 16) | (1u << 20) |                 // data_size=4B, pad_enable
                (pad_int_enc << 22) | (pad_amt_enc << 25));
  g1[1] = (int)((td0 & 0xFFFFu) << 16);                   // tensor_dim0[15:0]
  g1[2] = (int)((td0 >> 16) | ((td1 & 0xFFFFu) << 16));   // dim0[31:16]|dim1[15:0]
  g1[3] = (int)((td1 >> 16) | (tile_w_u32 << 16));        // dim1[31:16]|tile_dim0
  g1[4] = (int)(tile_h & 0xFFFFu);                        // tile_dim1; tile_dim2=0
  g1[5] = (int)stride_u32;                                // tensor_dim0_stride[31:0]
  g1[6] = 0;
  g1[7] = 0;
  i32x4 gz = {0, 0, 0, 0};
#if __has_include(<hip/amd_detail/amd_gfx1250_TDM.h>)
  i32x8 gz8 = {0, 0, 0, 0, 0, 0, 0, 0};
  __builtin_amdgcn_tensor_load_to_lds(g0, g1, gz, gz, gz8, 0);
#else
  __builtin_amdgcn_tensor_load_to_lds(g0, g1, gz, gz, 0);
#endif
}

__device__ __forceinline__ void tdm_wait() {
#if __has_builtin(__builtin_amdgcn_s_wait_tensorcnt)
  __builtin_amdgcn_s_wait_tensorcnt(0);
#else
  asm volatile("s_wait_tensorcnt 0x0" ::: "memory");
#endif
}
#endif  // HAVE_TDM

// ---------------------------------------------------------------------------
// f32 -> bf16 conversion (vectorized 4-wide)
// ---------------------------------------------------------------------------
__global__ void __launch_bounds__(256) cvt_bf16(const float* __restrict__ in,
                                                unsigned short* __restrict__ out,
                                                int n) {
  int i = (blockIdx.x * 256 + threadIdx.x) * 4;
  if (i + 3 < n) {
    float4 v = *(const float4*)(in + i);
    ushort4 r;
    r.x = f2bf(v.x); r.y = f2bf(v.y); r.z = f2bf(v.z); r.w = f2bf(v.w);
    *(ushort4*)(out + i) = r;
  }
}

// ---------------------------------------------------------------------------
// Out[M,N](bf16) = X[M,K](bf16) @ W[K,N](bf16) + bias[N](f32)
// M=4096, N=1024, K=1024.  Block tile 128x128x32, 8 waves (2x4).
// Double-buffered: next k-slice global loads issue before the 8 WMMAs.
// B staged TRANSPOSED ([n][k]) so all fragment reads are contiguous b128s.
// ---------------------------------------------------------------------------
__global__ void __launch_bounds__(256) gemm_bias(
    const unsigned short* __restrict__ X,
    const unsigned short* __restrict__ W,
    const float* __restrict__ bias,
    unsigned short* __restrict__ Out) {
  const int Kdim = C_DIM, Ndim = HD;
  __shared__ unsigned short As[2][128][40];   // [m][k] 128 x 32 (+8 pad)
  __shared__ unsigned short Bst[2][128][40];  // [n][k] 128 x 32 (+8 pad)

  const int tid  = threadIdx.x;
  const int wv   = tid >> 5;
  const int lane = tid & 31;
  const int lrow = lane & 15;
  const int hi   = lane >> 4;
  const int mw   = wv >> 2;                  // 0..1
  const int nw   = wv & 3;                   // 0..3
  const int m0   = blockIdx.y * 128;
  const int n0   = blockIdx.x * 128;

  union U { uint2 u; unsigned short s[4]; };

  const v8f vzero = {0.f, 0.f, 0.f, 0.f, 0.f, 0.f, 0.f, 0.f};
  v8f acc[4][2];
  for (int s = 0; s < 4; ++s)
    for (int u = 0; u < 2; ++u) acc[s][u] = vzero;

  // staging assignments (fixed per thread)
  const int b_nl = (tid & 31) * 4;                        // B: n-quad (0..124)
  const int b_k0 = (tid >> 5) * 4;                        // B: k-group (0..28)

  // ---- prologue: stage k-slice 0 into buffer 0 ----
  {
    uint2 ar[4];
    U br[4];
#pragma unroll
    for (int k = 0; k < 4; ++k) {
      int idx = tid + k * 256;
      int row = idx >> 3, v4 = (idx & 7) * 4;
      ar[k] = *(const uint2*)&X[(long)(m0 + row) * Kdim + v4];
    }
#pragma unroll
    for (int kk = 0; kk < 4; ++kk)
      br[kk].u = *(const uint2*)&W[(long)(b_k0 + kk) * Ndim + n0 + b_nl];
#pragma unroll
    for (int k = 0; k < 4; ++k) {
      int idx = tid + k * 256;
      int row = idx >> 3, v4 = (idx & 7) * 4;
      *(uint2*)&As[0][row][v4] = ar[k];
    }
#pragma unroll
    for (int nn = 0; nn < 4; ++nn) {
      U o;
      o.s[0] = br[0].s[nn]; o.s[1] = br[1].s[nn];
      o.s[2] = br[2].s[nn]; o.s[3] = br[3].s[nn];
      *(uint2*)&Bst[0][b_nl + nn][b_k0] = o.u;
    }
  }
  __syncthreads();

  for (int kt = 0; kt < Kdim; kt += 32) {
    const int cur = (kt >> 5) & 1;
    const int nxt = cur ^ 1;
    const bool have_next = (kt + 32) < Kdim;

    // -------- issue next-slice global loads (overlap with WMMAs) --------
    uint2 ar[4];
    U br[4];
    if (have_next) {
#pragma unroll
      for (int k = 0; k < 4; ++k) {
        int idx = tid + k * 256;
        int row = idx >> 3, v4 = (idx & 7) * 4;
        ar[k] = *(const uint2*)&X[(long)(m0 + row) * Kdim + kt + 32 + v4];
      }
#pragma unroll
      for (int kk = 0; kk < 4; ++kk)
        br[kk].u = *(const uint2*)&W[(long)(kt + 32 + b_k0 + kk) * Ndim + n0 + b_nl];
    }

    // -------- compute on current buffer --------
    v16bf af[4];
#pragma unroll
    for (int s = 0; s < 4; ++s)
#pragma unroll
      for (int j = 0; j < 16; ++j)
        af[s][j] = bits2bf(As[cur][mw * 64 + s * 16 + lrow][klocal(j, hi)]);
    v16bf bfr[2];
#pragma unroll
    for (int u = 0; u < 2; ++u)
#pragma unroll
      for (int j = 0; j < 16; ++j)
        bfr[u][j] = bits2bf(Bst[cur][nw * 32 + u * 16 + lrow][klocal(j, hi)]);
#pragma unroll
    for (int s = 0; s < 4; ++s)
#pragma unroll
      for (int u = 0; u < 2; ++u)
        acc[s][u] = wmma_bf16(af[s], bfr[u], acc[s][u]);

    // -------- publish next buffer --------
    if (have_next) {
#pragma unroll
      for (int k = 0; k < 4; ++k) {
        int idx = tid + k * 256;
        int row = idx >> 3, v4 = (idx & 7) * 4;
        *(uint2*)&As[nxt][row][v4] = ar[k];
      }
#pragma unroll
      for (int nn = 0; nn < 4; ++nn) {
        U o;
        o.s[0] = br[0].s[nn]; o.s[1] = br[1].s[nn];
        o.s[2] = br[2].s[nn]; o.s[3] = br[3].s[nn];
        *(uint2*)&Bst[nxt][b_nl + nn][b_k0] = o.u;
      }
    }
    __syncthreads();
  }

  // C/D layout: VGPR i -> M = i + hi*8 ; lane low nibble -> N
  float bsv[2];
#pragma unroll
  for (int u = 0; u < 2; ++u)
    bsv[u] = bias[n0 + nw * 32 + u * 16 + lrow];
#pragma unroll
  for (int s = 0; s < 4; ++s)
#pragma unroll
    for (int u = 0; u < 2; ++u)
#pragma unroll
      for (int i = 0; i < 8; ++i) {
        int m = m0 + mw * 64 + s * 16 + i + hi * 8;
        int n = n0 + nw * 32 + u * 16 + lrow;
        Out[(long)m * Ndim + n] = f2bf(acc[s][u][i] + bsv[u]);
      }
}

// ---------------------------------------------------------------------------
// Fused attention.  Per (b,h): Q',K',V' are D x N matrices (row stride F_LEN,
// contiguous last index).  S = Q'^T K' * alpha (+mask), online softmax over t,
// O[f][d] += P[f][t] * V'[d][t].  Output written as D x F.
// One block = (b, h, 128-query tile); 8 waves, each owns 16 queries.
// Double-buffered T loop: V(tile i+1) via TDM + K(tile i+1) register loads
// issue before the WMMAs of tile i; one barrier per iteration.
// ---------------------------------------------------------------------------
__global__ void __launch_bounds__(256) attn_kernel(
    const unsigned short* __restrict__ Qb,
    const unsigned short* __restrict__ Kb,
    const unsigned short* __restrict__ Vb,
    const float* __restrict__ mask,
    float* __restrict__ out) {
  __shared__ unsigned short Qs[64][136];        // [d][f_local] (+8 pad)
  __shared__ unsigned short Kst[2][64][72];     // [t_local][d] (+8 pad) TRANSPOSED
  __shared__ unsigned short Vs[2][64][72];      // [d][t_local] (+8 pad)
  __shared__ unsigned short Ps[8][16][72];      // per-wave P tile [f][t]

  const int tid  = threadIdx.x;
  const int w    = tid >> 5;
  const int lane = tid & 31;
  const int lrow = lane & 15;
  const int hi   = lane >> 4;

  const int b  = blockIdx.z;
  const int h  = blockIdx.y;
  const int f0 = blockIdx.x * 128;

  const long base = (long)b * BATCH_STRIDE + (long)h * (D_HEAD * F_LEN);

  union U { uint2 u; unsigned short s[4]; };

  // staging assignment for the K transpose (fixed per thread):
  const int k_tl = (tid & 15) * 4;           // t-quad (coalesced across lanes)
  const int k_d0 = (tid >> 4) * 4;           // d-group

  // ---- prologue: stage Q (64 x 128) and tile 0 of K/V under one barrier ----
#ifdef HAVE_TDM
  if (w == 0) {
    tdm_load_2d(&Qb[base + f0], lds_addr_of(&Qs[0][0]),
                /*tile_w_u32=*/64, /*tile_h=*/64, /*stride_u32=*/F_LEN / 2,
                /*pad: 64 dwords->enc5, 4 dwords->enc3*/ 5, 3);
    tdm_load_2d(&Vb[base], lds_addr_of(&Vs[0][0][0]),
                /*tile_w_u32=*/32, /*tile_h=*/64, /*stride_u32=*/T_LEN / 2,
                /*pad: 32 dwords->enc4, 4 dwords->enc3*/ 4, 3);
  }
#else
  {
    uint2 qr[8], vr[4];
#pragma unroll
    for (int k = 0; k < 8; ++k) {
      int idx = tid + k * 256;
      int row = idx >> 5, v4 = (idx & 31) * 4;
      qr[k] = *(const uint2*)&Qb[base + (long)row * F_LEN + f0 + v4];
    }
#pragma unroll
    for (int k = 0; k < 4; ++k) {
      int idx = tid + k * 256;
      int row = idx >> 4, v4 = (idx & 15) * 4;
      vr[k] = *(const uint2*)&Vb[base + (long)row * T_LEN + v4];
    }
#pragma unroll
    for (int k = 0; k < 8; ++k) {
      int idx = tid + k * 256;
      int row = idx >> 5, v4 = (idx & 31) * 4;
      *(uint2*)&Qs[row][v4] = qr[k];
    }
#pragma unroll
    for (int k = 0; k < 4; ++k) {
      int idx = tid + k * 256;
      int row = idx >> 4, v4 = (idx & 15) * 4;
      *(uint2*)&Vs[0][row][v4] = vr[k];
    }
  }
#endif
  {
    U kr[4];
#pragma unroll
    for (int dd = 0; dd < 4; ++dd)
      kr[dd].u = *(const uint2*)&Kb[base + (long)(k_d0 + dd) * T_LEN + k_tl];
#pragma unroll
    for (int tt = 0; tt < 4; ++tt) {
      U o2;
      o2.s[0] = kr[0].s[tt]; o2.s[1] = kr[1].s[tt];
      o2.s[2] = kr[2].s[tt]; o2.s[3] = kr[3].s[tt];
      *(uint2*)&Kst[0][k_tl + tt][k_d0] = o2.u;
    }
  }
#ifdef HAVE_TDM
  if (w == 0) tdm_wait();
#endif
  __syncthreads();

  // A fragment for S-gemm: M=f (16 rows of this wave), K=d (two 32-chunks)
  v16bf aq[2];
#pragma unroll
  for (int kk = 0; kk < 2; ++kk)
#pragma unroll
    for (int j = 0; j < 16; ++j)
      aq[kk][j] = bits2bf(Qs[kk * 32 + klocal(j, hi)][w * 16 + lrow]);

  const v8f vzero = {0.f, 0.f, 0.f, 0.f, 0.f, 0.f, 0.f, 0.f};
  v8f o[4];
  for (int nd = 0; nd < 4; ++nd) o[nd] = vzero;
  float row_max[8], row_sum[8];
  for (int i = 0; i < 8; ++i) { row_max[i] = -1e30f; row_sum[i] = 0.f; }

  for (int t0 = 0; t0 < T_LEN; t0 += 64) {
    const int cur = (t0 >> 6) & 1;
    const int nxt = cur ^ 1;
    const bool have_next = (t0 + 64) < T_LEN;

    // -------- issue next-tile loads (overlap with this tile's WMMAs) --------
    U kr[4];
#ifndef HAVE_TDM
    uint2 vr[4];
#endif
    if (have_next) {
      const int tn = t0 + 64;
#ifdef HAVE_TDM
      if (w == 0)
        tdm_load_2d(&Vb[base + tn], lds_addr_of(&Vs[nxt][0][0]),
                    32, 64, T_LEN / 2, 4, 3);
#else
#pragma unroll
      for (int k = 0; k < 4; ++k) {
        int idx = tid + k * 256;
        int row = idx >> 4, v4 = (idx & 15) * 4;
        vr[k] = *(const uint2*)&Vb[base + (long)row * T_LEN + tn + v4];
      }
#endif
#pragma unroll
      for (int dd = 0; dd < 4; ++dd)
        kr[dd].u = *(const uint2*)&Kb[base + (long)(k_d0 + dd) * T_LEN + tn + k_tl];
      if (t0 + 128 < T_LEN) {
        __builtin_prefetch(&Kb[base + (long)k_d0 * T_LEN + t0 + 128 + k_tl], 0, 1);
        __builtin_prefetch(&Vb[base + (long)k_d0 * T_LEN + t0 + 128 + k_tl], 0, 1);
      }
    }

    // ---- S tile: 16 (f) x 64 (t), K-dim = d = 64 (two WMMA k-steps) ----
    v8f sf[4];
#pragma unroll
    for (int nt = 0; nt < 4; ++nt) {
      v8f c = vzero;
#pragma unroll
      for (int kk = 0; kk < 2; ++kk) {
        v16bf bb;
#pragma unroll
        for (int j = 0; j < 16; ++j)
          bb[j] = bits2bf(Kst[cur][nt * 16 + lrow][kk * 32 + klocal(j, hi)]);
        c = wmma_bf16(aq[kk], bb, c);
      }
      sf[nt] = c;
    }

    // ---- alpha + mask ----
    const long mbase = (long)b * MASK_BSTRIDE;
#pragma unroll
    for (int nt = 0; nt < 4; ++nt)
#pragma unroll
      for (int i = 0; i < 8; ++i) {
        int fg = f0 + w * 16 + i + hi * 8;
        int tg = t0 + nt * 16 + lrow;
        float mv = mask[mbase + (long)fg * T_LEN + tg];
        sf[nt][i] = sf[nt][i] * 0.125f + (1.0f - mv) * (-100000.0f);
      }

    // ---- online softmax (row = fixed f; 16 t's per half-wave, 4 tiles) ----
    float scale[8];
#pragma unroll
    for (int i = 0; i < 8; ++i) {
      float m = fmaxf(fmaxf(sf[0][i], sf[1][i]), fmaxf(sf[2][i], sf[3][i]));
      m = fmaxf(m, __shfl_xor(m, 1, 32));
      m = fmaxf(m, __shfl_xor(m, 2, 32));
      m = fmaxf(m, __shfl_xor(m, 4, 32));
      m = fmaxf(m, __shfl_xor(m, 8, 32));
      float mn = fmaxf(row_max[i], m);
      scale[i] = __expf(row_max[i] - mn);
      row_max[i] = mn;
      float ls = 0.f;
#pragma unroll
      for (int nt = 0; nt < 4; ++nt) {
        float p = __expf(sf[nt][i] - mn);
        Ps[w][i + hi * 8][nt * 16 + lrow] = f2bf(p);
        ls += p;
      }
      ls += __shfl_xor(ls, 1, 32);
      ls += __shfl_xor(ls, 2, 32);
      ls += __shfl_xor(ls, 4, 32);
      ls += __shfl_xor(ls, 8, 32);
      row_sum[i] = row_sum[i] * scale[i] + ls;
#pragma unroll
      for (int nd = 0; nd < 4; ++nd) o[nd][i] *= scale[i];
    }

    // ---- O += P @ V'^T : M=f, N=d (4 tiles), K=t (two 32-chunks) ----
#pragma unroll
    for (int kk = 0; kk < 2; ++kk) {
      v16bf ap;
#pragma unroll
      for (int j = 0; j < 16; ++j)
        ap[j] = bits2bf(Ps[w][lrow][kk * 32 + klocal(j, hi)]);
#pragma unroll
      for (int nd = 0; nd < 4; ++nd) {
        v16bf bb;
#pragma unroll
        for (int j = 0; j < 16; ++j)
          bb[j] = bits2bf(Vs[cur][nd * 16 + lrow][kk * 32 + klocal(j, hi)]);
        o[nd] = wmma_bf16(ap, bb, o[nd]);
      }
    }

    // -------- publish next buffer --------
    if (have_next) {
#pragma unroll
      for (int tt = 0; tt < 4; ++tt) {
        U o2;
        o2.s[0] = kr[0].s[tt]; o2.s[1] = kr[1].s[tt];
        o2.s[2] = kr[2].s[tt]; o2.s[3] = kr[3].s[tt];
        *(uint2*)&Kst[nxt][k_tl + tt][k_d0] = o2.u;
      }
#ifndef HAVE_TDM
#pragma unroll
      for (int k = 0; k < 4; ++k) {
        int idx = tid + k * 256;
        int row = idx >> 4, v4 = (idx & 15) * 4;
        *(uint2*)&Vs[nxt][row][v4] = vr[k];
      }
#else
      if (w == 0) tdm_wait();
#endif
    }
    __syncthreads();
  }

  // ---- normalize + write out[b, h*64+d, f] ----
#pragma unroll
  for (int i = 0; i < 8; ++i) {
    float inv = 1.0f / row_sum[i];
    int fg = f0 + w * 16 + i + hi * 8;
#pragma unroll
    for (int nd = 0; nd < 4; ++nd) {
      int d = nd * 16 + lrow;
      out[(long)b * BATCH_STRIDE + (long)(h * D_HEAD + d) * F_LEN + fg] =
          o[nd][i] * inv;
    }
  }
}

// ---------------------------------------------------------------------------
// Launch
// ---------------------------------------------------------------------------
extern "C" void kernel_launch(void* const* d_in, const int* in_sizes, int n_in,
                              void* d_out, int out_size, void* d_ws, size_t ws_size,
                              hipStream_t stream) {
  const float* from = (const float*)d_in[0];
  const float* to   = (const float*)d_in[1];
  const float* mask = (const float*)d_in[2];
  const float* Wq   = (const float*)d_in[3];
  const float* bq   = (const float*)d_in[4];
  const float* Wk   = (const float*)d_in[5];
  const float* bk   = (const float*)d_in[6];
  const float* Wv   = (const float*)d_in[7];
  const float* bv   = (const float*)d_in[8];
  float* out = (float*)d_out;

  char* ws = (char*)d_ws;
  const size_t ACT = (size_t)B_SZ * F_LEN * C_DIM;      // 4194304 elems
  const size_t WEL = (size_t)C_DIM * HD;                // 1048576 elems
  unsigned short* Xbf = (unsigned short*)(ws);
  unsigned short* Ybf = (unsigned short*)(ws + 2 * ACT);
  unsigned short* Wqb = (unsigned short*)(ws + 4 * ACT);
  unsigned short* Wkb = (unsigned short*)(ws + 4 * ACT + 2 * WEL);
  unsigned short* Wvb = (unsigned short*)(ws + 4 * ACT + 4 * WEL);
  unsigned short* Qb  = (unsigned short*)(ws + 4 * ACT + 6 * WEL);
  unsigned short* Kb  = (unsigned short*)(ws + 6 * ACT + 6 * WEL);
  unsigned short* Vb  = (unsigned short*)(ws + 8 * ACT + 6 * WEL);

  cvt_bf16<<<dim3((unsigned)(ACT / 1024)), 256, 0, stream>>>(from, Xbf, (int)ACT);
  cvt_bf16<<<dim3((unsigned)(ACT / 1024)), 256, 0, stream>>>(to,   Ybf, (int)ACT);
  cvt_bf16<<<dim3((unsigned)(WEL / 1024)), 256, 0, stream>>>(Wq, Wqb, (int)WEL);
  cvt_bf16<<<dim3((unsigned)(WEL / 1024)), 256, 0, stream>>>(Wk, Wkb, (int)WEL);
  cvt_bf16<<<dim3((unsigned)(WEL / 1024)), 256, 0, stream>>>(Wv, Wvb, (int)WEL);

  dim3 ggrid(HD / 128, (B_SZ * F_LEN) / 128);            // (8, 32)
  gemm_bias<<<ggrid, 256, 0, stream>>>(Xbf, Wqb, bq, Qb);
  gemm_bias<<<ggrid, 256, 0, stream>>>(Ybf, Wkb, bk, Kb);
  gemm_bias<<<ggrid, 256, 0, stream>>>(Ybf, Wvb, bv, Vb);

  dim3 agrid(F_LEN / 128, NH, B_SZ);                     // (16, 16, 2)
  attn_kernel<<<agrid, 256, 0, stream>>>(Qb, Kb, Vb, mask, out);
}
